// RNNModel_31001073943189
// MI455X (gfx1250) — matrix operand order repaired and checked by reading, not compiled, not run on hardware
//
#include <hip/hip_runtime.h>
#include <hip/hip_bf16.h>

// ---------------------------------------------------------------------------
// GRU scan for MI455X (gfx1250, wave32, WMMA).
// B=128, T=1024, D=H=256.
//
// gru_pack: repack Wi and [Wh_rz|Wh_n] into bf16 WMMA-B-fragment layout in
//   workspace (768 KB, L2-resident).
// gru_scan: 8 persistent workgroups (one per 16-row batch tile), 8 waves.
//   h stays in LDS for all 1024 steps. Per step each wave runs a 16-stage
//   software pipeline (96 v_wmma_f32_16x16x32_bf16) with next-stage B-frags
//   loaded via global_load_b128 (addrspace(1)) and A-frags via ds_load_b128
//   while the current stage's WMMAs execute; then the fused GRU elementwise
//   update in registers, writing ys with non-temporal stores.
// ---------------------------------------------------------------------------

typedef __attribute__((ext_vector_type(16))) __bf16 bfx16;
typedef __attribute__((ext_vector_type(8)))  __bf16 bfx8;
typedef __attribute__((ext_vector_type(8)))  float  v8f;

typedef __bf16 __attribute__((address_space(1))) gbf;        // global bf16
typedef const bfx16 __attribute__((address_space(1)))* gbfx16p;

#define GRU_B 128
#define GRU_T 1024
#define GRU_D 256
#define GRU_H 256
#define PACK_PER_MAT (48 * 8 * 32 * 16)   // nt * kt * lane * k  = 196608 elems
#define APITCH 264                         // row pitch (528B) -> no LDS bank conflicts

__device__ __forceinline__ v8f wmma_bf16(bfx16 a, bfx16 b, v8f c) {
    // (neg_a, A, neg_b, B, c_mod, C, reuse_a, reuse_b)
    return __builtin_amdgcn_wmma_f32_16x16x32_bf16(false, a, false, b,
                                                   (short)0, c, false, false);
}

// B-matrix fragment (global, addrspace(1) -> global_load_b128):
// lane holds col = 16*nt + (lane&15), K = 32*kt + (lane>=16?8:0) + {0..7,16..23}.
__device__ __forceinline__ bfx16 bfrag_g(const gbf* W, int nt, int kt, int lane) {
    return *(gbfx16p)(W + ((((nt << 3) + kt) << 5) + lane) * 16);
}

// A-matrix fragment from LDS row (rowbase = &A[lane&15][0]):
// lanes 0-15 hold K = 32kt+{0..7,16..23}, lanes 16-31 hold K = 32kt+{8..15,24..31}
__device__ __forceinline__ bfx16 afrag(const __bf16* rowbase, int kt, int lhalf) {
    const bfx8 lo = *(const bfx8*)(rowbase + 32 * kt + lhalf * 8);
    const bfx8 hi = *(const bfx8*)(rowbase + 32 * kt + lhalf * 8 + 16);
    return __builtin_shufflevector(lo, hi, 0, 1, 2, 3, 4, 5, 6, 7,
                                           8, 9, 10, 11, 12, 13, 14, 15);
}

__device__ __forceinline__ float sigm_f(float x) {
    return 1.0f / (1.0f + __expf(-x));
}
__device__ __forceinline__ float tanh_f(float x) {
    float e = __expf(2.0f * x);
    return (e - 1.0f) / (e + 1.0f);
}

// ---------------------------------------------------------------------------
// Pack weights to bf16 fragment layout.
//   mat 0: xW  from Wi      [256 x 768]   (cols: ir | iz | in)
//   mat 1: hW  from Wh_rz   [256 x 512] and Wh_n [256 x 256] (cols: hr | hz | hn)
// packed[mat][nt][kt][lane][kk] = W[32*kt + (lane>>4)*8 + (kk&7) + (kk>>3)*16][16*nt + (lane&15)]
// ---------------------------------------------------------------------------
__global__ void gru_pack(const float* __restrict__ Wi,
                         const float* __restrict__ Wh_rz,
                         const float* __restrict__ Wh_n,
                         __bf16* __restrict__ ws) {
    int idx = blockIdx.x * blockDim.x + threadIdx.x;
    if (idx >= 2 * PACK_PER_MAT) return;
    int mat = idx / PACK_PER_MAT;
    int e   = idx % PACK_PER_MAT;
    int kk = e & 15;
    int L  = (e >> 4) & 31;
    int kt = (e >> 9) & 7;
    int nt = e >> 12;                       // 0..47
    int K   = 32 * kt + ((L >> 4) << 3) + (kk & 7) + ((kk >> 3) << 4);
    int col = (nt << 4) + (L & 15);
    float v;
    if (mat == 0) {
        v = Wi[K * 768 + col];
    } else {
        v = (col < 512) ? Wh_rz[K * 512 + col] : Wh_n[K * 256 + (col - 512)];
    }
    ws[idx] = (__bf16)v;
}

// ---------------------------------------------------------------------------
// Persistent GRU scan. grid = B/16 = 8, block = 256 (8 waves).
// ---------------------------------------------------------------------------
__global__ __launch_bounds__(256, 1)
__attribute__((amdgpu_waves_per_eu(2)))
void gru_scan(const float* __restrict__ obs,
              const unsigned char* __restrict__ done,
              const float* __restrict__ h0,
              const float* __restrict__ bi,
              const float* __restrict__ bhn,
              const __bf16* __restrict__ ws,
              float* __restrict__ out_final,
              float* __restrict__ out_ys) {
    __shared__ __bf16 xA[16][APITCH];        // obs tile, bf16
    __shared__ __bf16 hA[16][APITCH];        // h tile, bf16 (masked)
    __shared__ float  hS[16][APITCH];        // h state, f32 (masked)

    const int tid   = threadIdx.x;
    const int b0    = blockIdx.x * 16;
    const int lane  = tid & 31;
    const int wv    = tid >> 5;              // wave 0..7
    const int lhalf = lane >> 4;
    const int lm    = lane & 15;

    const int mrow = tid >> 4;               // 0..15 (cooperative row)
    const int ccol = (tid & 15) * 16;        // 16-col chunk per thread

    // init h state
    #pragma unroll
    for (int j = 0; j < 16; j++)
        hS[mrow][ccol + j] = h0[(b0 + mrow) * GRU_H + ccol + j];
    __syncthreads();

    // per-wave column assignment: 32 hidden columns starting at cb
    const int cb = wv * 32;
    const int c0l = cb + lm, c1l = c0l + 16;
    const float birA[2] = { bi[c0l],        bi[c1l]        };
    const float bizA[2] = { bi[256 + c0l],  bi[256 + c1l]  };
    const float binA[2] = { bi[512 + c0l],  bi[512 + c1l]  };
    const float bhA[2]  = { bhn[c0l],       bhn[c1l]       };

    // tile order per stage: r0 r1 z0 z1 n0 n1
    const int nts[6] = { 2 * wv, 2 * wv + 1,
                         16 + 2 * wv, 17 + 2 * wv,
                         32 + 2 * wv, 33 + 2 * wv };

    for (int t = 0; t < GRU_T; t++) {
        // Opaque per-iteration offset: blocks hoisting of the fragment loads
        // out of the t-loop (prevents 768-VGPR spill). Pointer keeps
        // addrspace(1) so loads lower to global_load_b128 (LOADcnt only).
        unsigned long woff = 0;
        asm volatile("" : "+s"(woff));
        const gbf* xWg = (const gbf*)(unsigned long)(ws) + woff;
        const gbf* hWg = xWg + PACK_PER_MAT;

        // ---- phase 1: stage obs tile + masked h (f32 + bf16) into LDS ----
        {
            const unsigned char dflag = done[(b0 + mrow) * GRU_T + t];
            const size_t obase = ((size_t)(b0 + mrow) * GRU_T + t) * GRU_D + ccol;
            const float4* op = (const float4*)(obs + obase);
            #pragma unroll
            for (int q = 0; q < 4; q++) {
                float4 x = op[q];
                xA[mrow][ccol + 4 * q + 0] = (__bf16)x.x;
                xA[mrow][ccol + 4 * q + 1] = (__bf16)x.y;
                xA[mrow][ccol + 4 * q + 2] = (__bf16)x.z;
                xA[mrow][ccol + 4 * q + 3] = (__bf16)x.w;
            }
            #pragma unroll
            for (int j = 0; j < 16; j++) {
                float hv = dflag ? 0.0f : hS[mrow][ccol + j];
                hS[mrow][ccol + j] = hv;
                hA[mrow][ccol + j] = (__bf16)hv;
            }
            if (t + 1 < GRU_T)   // warm L2 for next step's obs rows
                __builtin_prefetch(obs + obase + GRU_D, 0, 0);
        }
        __syncthreads();

        // ---- phase 2: 16-stage software-pipelined WMMA (fp32 accumulate) --
        // stages 0..7  : A = obs tile,  B = xW, accumulate R/Z/XN
        // stages 8..15 : A = h tile,    B = hW, accumulate R/Z/HN
        v8f accR[2]  = { v8f{}, v8f{} };
        v8f accZ[2]  = { v8f{}, v8f{} };
        v8f accXN[2] = { v8f{}, v8f{} };
        v8f accHN[2] = { v8f{}, v8f{} };

        bfx16 aC = afrag(&xA[lm][0], 0, lhalf);
        bfx16 bC[6];
        #pragma unroll
        for (int j = 0; j < 6; j++) bC[j] = bfrag_g(xWg, nts[j], 0, lane);

        #pragma unroll
        for (int s = 0; s < 16; s++) {
            bfx16 aN;
            bfx16 bN[6];
            if (s < 15) {                       // prefetch next stage
                const int kt = (s + 1) & 7;
                const __bf16* Asrc = (s + 1 < 8) ? &xA[lm][0] : &hA[lm][0];
                const gbf*    Wsrc = (s + 1 < 8) ? xWg : hWg;
                aN = afrag(Asrc, kt, lhalf);
                #pragma unroll
                for (int j = 0; j < 6; j++) bN[j] = bfrag_g(Wsrc, nts[j], kt, lane);
            }
            accR[0] = wmma_bf16(aC, bC[0], accR[0]);
            accR[1] = wmma_bf16(aC, bC[1], accR[1]);
            accZ[0] = wmma_bf16(aC, bC[2], accZ[0]);
            accZ[1] = wmma_bf16(aC, bC[3], accZ[1]);
            if (s < 8) {
                accXN[0] = wmma_bf16(aC, bC[4], accXN[0]);
                accXN[1] = wmma_bf16(aC, bC[5], accXN[1]);
            } else {
                accHN[0] = wmma_bf16(aC, bC[4], accHN[0]);
                accHN[1] = wmma_bf16(aC, bC[5], accHN[1]);
            }
            if (s < 15) {
                aC = aN;
                #pragma unroll
                for (int j = 0; j < 6; j++) bC[j] = bN[j];
            }
        }

        // ---- phase 3: fused GRU elementwise update in registers ----
        // C/D layout: VGPR i -> row (i + 8*lhalf), lane%16 -> column.
        #pragma unroll
        for (int s = 0; s < 2; s++) {
            const int col = cb + 16 * s + lm;
            const float bir = birA[s], biz = bizA[s], bin = binA[s], bh = bhA[s];
            #pragma unroll
            for (int i = 0; i < 8; i++) {
                const int mm = lhalf * 8 + i;
                float r = sigm_f(accR[s][i] + bir);
                float z = sigm_f(accZ[s][i] + biz);
                float n = tanh_f(accXN[s][i] + bin + r * (accHN[s][i] + bh));
                float hold = hS[mm][col];
                float hnew = (1.0f - z) * n + z * hold;
                hS[mm][col] = hnew;
                __builtin_nontemporal_store(
                    hnew, &out_ys[((size_t)(b0 + mm) * GRU_T + t) * GRU_H + col]);
            }
        }
        __syncthreads();
    }

    // final carry
    #pragma unroll
    for (int j = 0; j < 16; j++)
        out_final[(b0 + mrow) * GRU_H + ccol + j] = hS[mrow][ccol + j];
}

// ---------------------------------------------------------------------------
extern "C" void kernel_launch(void* const* d_in, const int* in_sizes, int n_in,
                              void* d_out, int out_size, void* d_ws, size_t ws_size,
                              hipStream_t stream) {
    const float*         obs   = (const float*)d_in[0];
    const unsigned char* done  = (const unsigned char*)d_in[1];  // jnp bool = 1 byte
    const float*         h0    = (const float*)d_in[2];
    const float*         Wi    = (const float*)d_in[3];
    const float*         bi    = (const float*)d_in[4];
    const float*         Wh_rz = (const float*)d_in[5];
    const float*         Wh_n  = (const float*)d_in[6];
    const float*         bh_n  = (const float*)d_in[7];

    __bf16* wsb = (__bf16*)d_ws;                 // 2*196608 bf16 = 768 KB
    float* out_final = (float*)d_out;            // [B,H] first (return order)
    float* out_ys    = out_final + GRU_B * GRU_H;

    gru_pack<<<(2 * PACK_PER_MAT + 255) / 256, 256, 0, stream>>>(Wi, Wh_rz, Wh_n, wsb);
    gru_scan<<<GRU_B / 16, 256, 0, stream>>>(obs, done, h0, bi, bh_n, wsb,
                                             out_final, out_ys);
}
